// ETAttention_16312285790820
// MI455X (gfx1250) — compile-verified, matrix operand order checked by reading, not run on hardware
//
#include <hip/hip_runtime.h>

// ET-Attention fused pipeline for gfx1250 (MI455X), wave32 + WMMA f16->f32,
// TDM double-buffered K/V staging in the flash-attention inner loop.
// B=4, T=2048, D=256, H=8, KQ=V=16.

typedef __attribute__((ext_vector_type(16))) _Float16 v16h;
typedef __attribute__((ext_vector_type(8)))  float    v8f;
typedef __attribute__((ext_vector_type(4)))  unsigned int u32x4;
typedef __attribute__((ext_vector_type(8)))  int      i32x8;
typedef __attribute__((ext_vector_type(4)))  int      i32x4;

#define B_   4
#define T_   2048
#define D_   256
#define H_   8
#define NT_  (T_ / 16)      // 128 16-row tiles per (b,h)
#define BT_  (B_ * T_)      // 8192

union FragH { v16h v; unsigned int u[8]; };

__device__ __forceinline__ unsigned int pkf16(float a, float b) {
  union { _Float16 h[2]; unsigned int u; } r;
  r.h[0] = (_Float16)a; r.h[1] = (_Float16)b;
  return r.u;
}

// K index of fragment dword j (0..7) for lane half hf, inside a 32-wide chunk.
// Same mapping for 16-bit A (lane=M) and B (lane=N) fragments, ISA 7.12.2.
__device__ __forceinline__ int kmap(int j, int hf) {
  return (j < 4) ? (hf * 8 + 2 * j) : (16 + hf * 8 + 2 * (j - 4));
}

__device__ __forceinline__ v8f wmma_f16(const FragH& a, const FragH& b, v8f c) {
  return __builtin_amdgcn_wmma_f32_16x16x32_f16(
      false, a.v, false, b.v, (short)0, c, false, false);
}

__device__ __forceinline__ float fast_rcp(float x) {
  return __builtin_amdgcn_rcpf(x);
}
__device__ __forceinline__ float silu_f(float x) {
  return x * fast_rcp(1.0f + __expf(-x));
}

__device__ __forceinline__ void setA(FragH& A, uint4 lo, uint4 hi) {
  A.u[0] = lo.x; A.u[1] = lo.y; A.u[2] = lo.z; A.u[3] = lo.w;
  A.u[4] = hi.x; A.u[5] = hi.y; A.u[6] = hi.z; A.u[7] = hi.w;
}
__device__ __forceinline__ void setHalf(FragH& A, uint4 lo) {
  A.u[0] = lo.x; A.u[1] = lo.y; A.u[2] = lo.z; A.u[3] = lo.w;
  A.u[4] = A.u[5] = A.u[6] = A.u[7] = 0;
}
// A-fragment from f16 LDS tile st[16][128]: two aligned b128 reads.
__device__ __forceinline__ void loadA_st16(FragH& A, const _Float16* st,
                                           int row, int kk, int hf) {
  const _Float16* p = st + row * 128 + kk * 32 + 8 * hf;
  setA(A, *(const uint4*)p, *(const uint4*)(p + 16));
}

// ---------------------------------------------------------------------------
// Tensor Data Mover: 1-D copy of `bytes` (mult of 4) global -> LDS.
// D# per CDNA5 ISA ch.8: group0 {count=1, lds_addr, global_addr, type=2},
// group1 {data_size=1B, tensor_dim0=bytes, tensor_dim1=1, tile_dim0=bytes,
//         tensor_dim0_stride=bytes}.  Tracked by TENSORcnt.
// ---------------------------------------------------------------------------
__device__ __forceinline__ void tdm_load_1d(const void* gsrc, unsigned lds_off,
                                            unsigned bytes) {
  unsigned long long ga = (unsigned long long)(size_t)gsrc;
  u32x4 g0;
  g0[0] = 1u;                                            // count=1 (valid)
  g0[1] = lds_off;                                       // lds_addr
  g0[2] = (unsigned)ga;                                  // global_addr[31:0]
  g0[3] = (unsigned)((ga >> 32) & 0x01FFFFFFu) | (2u << 30); // addr hi | type=2
  i32x8 g1;
  g1[0] = 0;                                             // wg_mask=0, 1B elems
  g1[1] = (int)(bytes << 16);                            // tensor_dim0[15:0]
  g1[2] = (int)((bytes >> 16) | (1u << 16));             // dim0 hi | tensor_dim1=1
  g1[3] = (int)(bytes << 16);                            // tile_dim0
  g1[4] = 0;                                             // tile_dim1/2 unused
  g1[5] = (int)bytes;                                    // tensor_dim0_stride lo
  g1[6] = 0; g1[7] = 0;
  i32x4 g2 = (i32x4)0, g3 = (i32x4)0;
#if defined(__clang_major__) && (__clang_major__ >= 23)
  i32x8 g4 = (i32x8)0;
  __builtin_amdgcn_tensor_load_to_lds(g0, g1, g2, g3, g4, 0);
#else
  __builtin_amdgcn_tensor_load_to_lds(g0, g1, g2, g3, 0);
#endif
}

// ---------------------------------------------------------------------------
// Kernel 0: pack f32 weight matrix W[K x N] into f16 B-fragment tiles:
// dst[((ntile*(K/32) + kk)*32 + lane)] = two uint4 (8 f16 pairs).
// ---------------------------------------------------------------------------
__global__ __launch_bounds__(256)
void et_packw_kernel(const float* __restrict__ W, int K, int N, int Npad,
                     uint4* __restrict__ dst)
{
  int tid = blockIdx.x * blockDim.x + threadIdx.x;
  int nkk = K >> 5;
  int total = (Npad >> 4) * nkk * 32;
  if (tid >= total) return;
  int lane = tid & 31, rest = tid >> 5;
  int kk = rest % nkk, ntile = rest / nkk;
  int hf = lane >> 4;
  int n = ntile * 16 + (lane & 15);
  unsigned int u[8];
  #pragma unroll
  for (int j = 0; j < 8; ++j) {
    int K0 = kk * 32 + kmap(j, hf);
    float w0 = (n < N) ? W[(size_t)K0 * N + n] : 0.0f;
    float w1 = (n < N) ? W[(size_t)(K0 + 1) * N + n] : 0.0f;
    u[j] = pkf16(w0, w1);
  }
  dst[(size_t)tid * 2]     = make_uint4(u[0], u[1], u[2], u[3]);
  dst[(size_t)tid * 2 + 1] = make_uint4(u[4], u[5], u[6], u[7]);
}

// ---------------------------------------------------------------------------
// Kernel 1: QKV + v_x MLPs, RoPE, fragment-layout store.
// One workgroup (8 waves) per 16-row time tile; wave w == head w.
// kv tile layout per (b,h,kt): 96 uint4 = [k-frag | v0-frag | v1-frag].
// ---------------------------------------------------------------------------
__global__ __launch_bounds__(256)
void et_qkv_kernel(const float* __restrict__ x, const float* __restrict__ h,
                   const float* __restrict__ qb1, const float* __restrict__ qb2,
                   const float* __restrict__ kb1, const float* __restrict__ kb2,
                   const float* __restrict__ vb1, const float* __restrict__ vb2,
                   const float* __restrict__ xb1, const float* __restrict__ xb2,
                   const uint4* __restrict__ pwq1, const uint4* __restrict__ pwq2,
                   const uint4* __restrict__ pwk1, const uint4* __restrict__ pwk2,
                   const uint4* __restrict__ pwv1, const uint4* __restrict__ pwv2,
                   const uint4* __restrict__ pwx1, const uint4* __restrict__ pwx2,
                   uint4* __restrict__ qfrag, uint4* __restrict__ kvfrag)
{
  __shared__ uint4     a_lds[8][32][2];   // A-fragments of h_m tile (f16)
  __shared__ _Float16  st16[16 * 128];    // f16 staging [row][col]
  __shared__ float     vx_lds[16 * 8];    // v_x tile [time][head]

  const int lane = threadIdx.x & 31;
  const int ww   = threadIdx.x >> 5;      // wave id == head id == 16-col group
  const int hf   = lane >> 4;
  const int col  = lane & 15;
  const int tile = blockIdx.x;            // 0..511
  const int b    = tile >> 7;
  const int kt   = tile & (NT_ - 1);
  const int t0   = kt * 16;
  const int c0   = ww * 16;
  const int n    = c0 + col;

  // ---- stage A fragments of h_m (16 rows x 256): wave w builds chunk w ----
  {
    unsigned int u[8];
    #pragma unroll
    for (int j = 0; j < 8; ++j) {
      int K = ww * 32 + kmap(j, hf);
      u[j] = pkf16(h[((size_t)b * D_ + K    ) * T_ + t0 + col],
                   h[((size_t)b * D_ + K + 1) * T_ + t0 + col]);
    }
    a_lds[ww][lane][0] = make_uint4(u[0], u[1], u[2], u[3]);
    a_lds[ww][lane][1] = make_uint4(u[4], u[5], u[6], u[7]);
  }
  __syncthreads();

  // ---------------- q and k branches ----------------
  for (int qk = 0; qk < 2; ++qk) {
    const uint4* W1 = qk ? pwk1 : pwq1;  const float* Bb1 = qk ? kb1 : qb1;
    const uint4* W2 = qk ? pwk2 : pwq2;  const float* Bb2 = qk ? kb2 : qb2;

    v8f acc = {};
    for (int kk = 0; kk < 8; ++kk) {                       // layer 1, K=256
      FragH A, Bf;
      setA(A, a_lds[kk][lane][0], a_lds[kk][lane][1]);
      size_t wi = ((size_t)(ww * 8 + kk) * 32 + lane) * 2;
      setA(Bf, W1[wi], W1[wi + 1]);
      acc = wmma_f16(A, Bf, acc);
    }
    {
      float b1 = Bb1[n];
      #pragma unroll
      for (int r = 0; r < 8; ++r)
        st16[(r + 8 * hf) * 128 + n] = (_Float16)silu_f(acc[r] + b1);
    }
    __syncthreads();

    v8f acc2 = {};
    for (int kk = 0; kk < 4; ++kk) {                       // layer 2, K=128
      FragH A, Bf;
      loadA_st16(A, st16, col, kk, hf);
      size_t wi = ((size_t)(ww * 4 + kk) * 32 + lane) * 2;
      setA(Bf, W2[wi], W2[wi + 1]);
      acc2 = wmma_f16(A, Bf, acc2);
    }
    __syncthreads();

    // RoPE (+ fold 1/sqrt(KQ) into q) -> st16 as f16 [time][feat]
    {
      float b2 = Bb2[n];
      const int   pairi = col >> 1;
      const float inv   = __powf(10000.0f, -(float)(2 * pairi) / 16.0f);
      const float qscal = qk ? 1.0f : 0.25f;
      #pragma unroll
      for (int r = 0; r < 8; ++r) {
        int   row = r + 8 * hf;
        float vv  = acc2[r] + b2;
        float pr  = __shfl_xor(vv, 1);
        float sn, cs;
        __sincosf((float)(t0 + row) * inv, &sn, &cs);
        float rot = (col & 1) ? (vv * cs + pr * sn) : (vv * cs - pr * sn);
        st16[row * 128 + n] = (_Float16)(rot * qscal);
      }
    }
    __syncthreads();

    // Gather fragment (lane = time, dwords = feat pairs) and store
    {
      const _Float16* p = st16 + col * 128 + c0 + 8 * hf;
      uint4 o = *(const uint4*)p;
      if (qk == 0)
        qfrag[((size_t)(b * H_ + ww) * NT_ + kt) * 32 + lane] = o;
      else
        kvfrag[((size_t)(b * H_ + ww) * NT_ + kt) * 96 + lane] = o;
    }
    __syncthreads();
  }

  // ---------------- v branch + extended value columns ----------------
  {
    v8f acc = {};
    for (int kk = 0; kk < 8; ++kk) {
      FragH A, Bf;
      setA(A, a_lds[kk][lane][0], a_lds[kk][lane][1]);
      size_t wi = ((size_t)(ww * 8 + kk) * 32 + lane) * 2;
      setA(Bf, pwv1[wi], pwv1[wi + 1]);
      acc = wmma_f16(A, Bf, acc);
    }
    {
      float b1 = vb1[n];
      #pragma unroll
      for (int r = 0; r < 8; ++r)
        st16[(r + 8 * hf) * 128 + n] = (_Float16)silu_f(acc[r] + b1);
    }
    __syncthreads();

    v8f acc2 = {};
    for (int kk = 0; kk < 4; ++kk) {
      FragH A, Bf;
      loadA_st16(A, st16, col, kk, hf);
      size_t wi = ((size_t)(ww * 4 + kk) * 32 + lane) * 2;
      setA(Bf, pwv2[wi], pwv2[wi + 1]);
      acc2 = wmma_f16(A, Bf, acc2);
    }
    float vc[8];
    {
      float b2 = vb2[n];
      #pragma unroll
      for (int r = 0; r < 8; ++r) vc[r] = acc2[r] + b2;
    }
    // v0-fragment comes straight out of the C layout (pairs of rows).
    kvfrag[((size_t)(b * H_ + ww) * NT_ + kt) * 96 + 32 + lane] =
        make_uint4(pkf16(vc[0], vc[1]), pkf16(vc[2], vc[3]),
                   pkf16(vc[4], vc[5]), pkf16(vc[6], vc[7]));
    __syncthreads();
    #pragma unroll
    for (int r = 0; r < 8; ++r)
      st16[(r + 8 * hf) * 128 + n] = (_Float16)vc[r];      // v as f16 [t][c]
    __syncthreads();

    // xm MLP layer 1: A = v (st16), B = packed xm_w1
    v8f accx = {};
    for (int kk = 0; kk < 4; ++kk) {
      FragH A, Bf;
      loadA_st16(A, st16, col, kk, hf);
      size_t wi = ((size_t)(ww * 4 + kk) * 32 + lane) * 2;
      setA(Bf, pwx1[wi], pwx1[wi + 1]);
      accx = wmma_f16(A, Bf, accx);
    }
    float bx1 = xb1[n];
    __syncthreads();
    #pragma unroll
    for (int r = 0; r < 8; ++r)
      st16[(r + 8 * hf) * 128 + n] = (_Float16)silu_f(accx[r] + bx1);
    __syncthreads();

    // xm MLP layer 2 (8 output cols; wave 0) -> v_x[time][head]
    if (ww == 0) {
      v8f a2 = {};
      for (int kk = 0; kk < 4; ++kk) {
        FragH A, Bf;
        loadA_st16(A, st16, col, kk, hf);
        size_t wi = ((size_t)kk * 32 + lane) * 2;
        setA(Bf, pwx2[wi], pwx2[wi + 1]);
        a2 = wmma_f16(A, Bf, a2);
      }
      if (col < 8) {
        float bb = xb2[col];
        #pragma unroll
        for (int r = 0; r < 8; ++r)
          vx_lds[(r + 8 * hf) * 8 + col] = a2[r] + bb;
      }
    }
    __syncthreads();

    // v1-fragment: extended cols 16..31 = [v_x, v_x*x0, v_x*x1, v_x*x2, 0..]
    {
      unsigned int o[4];
      #pragma unroll
      for (int j = 0; j < 4; ++j) {
        float e[2];
        #pragma unroll
        for (int ee = 0; ee < 2; ++ee) {
          int   tt = 2 * j + 8 * hf + ee;
          float vx = vx_lds[tt * 8 + ww];
          float val = 0.0f;
          if (col == 0)      val = vx;
          else if (col <= 3) val = vx * x[((size_t)b * 3 + (col - 1)) * T_ + t0 + tt];
          e[ee] = val;
        }
        o[j] = pkf16(e[0], e[1]);
      }
      kvfrag[((size_t)(b * H_ + ww) * NT_ + kt) * 96 + 64 + lane] =
          make_uint4(o[0], o[1], o[2], o[3]);
    }
  }
}

// ---------------------------------------------------------------------------
// Kernel 2: fused flash attention, extended V (20 live cols, padded to 32).
// 8 waves/block share one (b,h): K/V tiles staged to LDS by the Tensor Data
// Mover (double buffered, one barrier + s_wait_tensorcnt per tile).
// ---------------------------------------------------------------------------
__global__ __launch_bounds__(256)
void et_attn_kernel(const uint4* __restrict__ qfrag,
                    const uint4* __restrict__ kvfrag,
                    float* __restrict__ hupd, float* __restrict__ xacc)
{
  __shared__ uint4 kvbuf[2][96];           // [k | v0 | v1] fragments, 1.5 KB

  const int lane = threadIdx.x & 31;
  const int ww   = threadIdx.x >> 5;
  const int gid  = blockIdx.x * 8 + ww;    // 0..4095; 8 waves share one bh
  const int qt   = gid & (NT_ - 1);
  const int bh   = gid >> 7;
  const int hf   = lane >> 4;
  const int col  = lane & 15;

  const uint4* kvb = &kvfrag[(size_t)bh * NT_ * 96];

  FragH bq;
  setHalf(bq, qfrag[((size_t)bh * NT_ + qt) * 32 + lane]);

  if (ww == 0)
    tdm_load_1d(&kvb[0], (unsigned)(size_t)&kvbuf[0][0], 96 * 16);

  float m = -1.0e30f, lsum = 0.0f;
  v8f o0 = {}, o1 = {};

  for (int kt = 0; kt < NT_; ++kt) {
    const int cur = kt & 1;
    if (ww == 0)
      __builtin_amdgcn_s_wait_tensorcnt(0);       // tile kt landed in LDS
    __syncthreads();
    if (ww == 0 && kt + 1 < NT_)
      tdm_load_1d(&kvb[(size_t)(kt + 1) * 96],
                  (unsigned)(size_t)&kvbuf[cur ^ 1][0], 96 * 16);
    if (ww == 1 && kt + 2 < NT_)                  // keep TDM fed from L2
      __builtin_prefetch((const void*)&kvb[(size_t)(kt + 2) * 96 + lane * 3], 0, 0);

    FragH ak, bv0, bv1;
    setHalf(ak,  kvbuf[cur][lane]);
    setHalf(bv0, kvbuf[cur][32 + lane]);
    setHalf(bv1, kvbuf[cur][64 + lane]);

    v8f sT = {};
    sT = wmma_f16(ak, bq, sT);                    // sT[key, query]

    float tm = sT[0];
    #pragma unroll
    for (int r = 1; r < 8; ++r) tm = fmaxf(tm, sT[r]);
    tm = fmaxf(tm, __shfl_xor(tm, 16));           // combine key halves
    float mnew  = fmaxf(m, tm);
    float alpha = __expf(m - mnew);
    float p[8], tsum = 0.0f;
    #pragma unroll
    for (int r = 0; r < 8; ++r) { p[r] = __expf(sT[r] - mnew); tsum += p[r]; }
    tsum += __shfl_xor(tsum, 16);
    lsum = lsum * alpha + tsum;
    m = mnew;

    FragH ap;                                     // P packs straight into A
    ap.u[0] = pkf16(p[0], p[1]); ap.u[1] = pkf16(p[2], p[3]);
    ap.u[2] = pkf16(p[4], p[5]); ap.u[3] = pkf16(p[6], p[7]);
    ap.u[4] = ap.u[5] = ap.u[6] = ap.u[7] = 0;

    #pragma unroll
    for (int r = 0; r < 8; ++r) {                 // rescale running rows
      float ar = __shfl(alpha, r + 8 * hf);
      o0[r] *= ar; o1[r] *= ar;
    }
    o0 = wmma_f16(ap, bv0, o0);
    o1 = wmma_f16(ap, bv1, o1);
  }

  #pragma unroll
  for (int r = 0; r < 8; ++r) {
    int    qrow = r + 8 * hf;
    float  inv  = fast_rcp(__shfl(lsum, qrow));
    size_t t    = (size_t)qt * 16 + qrow;
    hupd[((size_t)bh * T_ + t) * 16 + col] = o0[r] * inv;
    if (col < 4)
      xacc[((size_t)bh * T_ + t) * 4 + col] = o1[r] * inv;  // gamma, wx0..2
  }
}

// ---------------------------------------------------------------------------
// Kernel 3: h_out = h + hupd_flat[8192x128] @ ho_w[128x256] + ho_b
// ---------------------------------------------------------------------------
__global__ __launch_bounds__(256)
void et_hout_kernel(const float* __restrict__ h, const float* __restrict__ hupd,
                    const uint4* __restrict__ pwho, const float* __restrict__ hob,
                    float* __restrict__ hout)
{
  const int lane = threadIdx.x & 31;
  const int ww   = threadIdx.x >> 5;
  const int hf   = lane >> 4;
  const int col  = lane & 15;
  const int tile = blockIdx.x;          // 0..511
  const int b    = tile >> 7;
  const int t0   = (tile & (NT_ - 1)) * 16;
  const int n0   = ww * 32;
  const int n    = n0 + col;

  v8f a0 = {}, a1 = {};
  for (int kk = 0; kk < 4; ++kk) {
    FragH A, B0, B1;
    #pragma unroll
    for (int j = 0; j < 8; ++j) {
      int K = kk * 32 + kmap(j, hf);            // even: pair stays in head
      int head = K >> 4, c = K & 15;
      const float* hp = &hupd[(((size_t)b * H_ + head) * T_ + t0 + col) * 16 + c];
      A.u[j] = pkf16(hp[0], hp[1]);
    }
    size_t w0 = ((size_t)((2 * ww) * 4 + kk) * 32 + lane) * 2;
    size_t w1 = ((size_t)((2 * ww + 1) * 4 + kk) * 32 + lane) * 2;
    setA(B0, pwho[w0], pwho[w0 + 1]);
    setA(B1, pwho[w1], pwho[w1 + 1]);
    a0 = wmma_f16(A, B0, a0);
    a1 = wmma_f16(A, B1, a1);
  }
  #pragma unroll
  for (int r = 0; r < 8; ++r) {
    int t  = t0 + r + 8 * hf;
    int d0 = n, d1 = n + 16;
    size_t i0 = ((size_t)b * D_ + d0) * T_ + t;
    size_t i1 = ((size_t)b * D_ + d1) * T_ + t;
    hout[i0] = h[i0] + a0[r] + hob[d0];
    hout[i1] = h[i1] + a1[r] + hob[d1];
  }
}

// ---------------------------------------------------------------------------
// Kernel 4: GVP vector branch -> x_out (scalar per (b,t))
// ---------------------------------------------------------------------------
__global__ __launch_bounds__(256)
void et_xout_kernel(const float* __restrict__ x, const float* __restrict__ xacc,
                    const float* __restrict__ Wh, const float* __restrict__ Wu,
                    float* __restrict__ xout)
{
  int idx = blockIdx.x * blockDim.x + threadIdx.x;
  if (idx >= BT_) return;
  int b = idx / T_, t = idx % T_;

  float xv[3];
  #pragma unroll
  for (int c = 0; c < 3; ++c) xv[c] = x[((size_t)b * 3 + c) * T_ + t];

  float Vu[3] = {0.0f, 0.0f, 0.0f};
  #pragma unroll
  for (int v = 0; v < H_; ++v) {
    float wc = 0.0f;                              // (Wh @ Wu)[v]
    #pragma unroll
    for (int hh = 0; hh < H_; ++hh) wc += Wh[v * H_ + hh] * Wu[hh];
    const float* xa = &xacc[(((size_t)b * H_ + v) * T_ + t) * 4];
    float gamma = xa[0];
    #pragma unroll
    for (int c = 0; c < 3; ++c)
      Vu[c] += (gamma * xv[c] - xa[1 + c]) * wc;
  }
  float gate = sqrtf(Vu[0] * Vu[0] + Vu[1] * Vu[1] + Vu[2] * Vu[2]);
  float sg   = fast_rcp(1.0f + __expf(-gate));
  #pragma unroll
  for (int c = 0; c < 3; ++c)
    xout[((size_t)b * 3 + c) * T_ + t] = xv[c] + sg * Vu[c];
}

// ---------------------------------------------------------------------------
extern "C" void kernel_launch(void* const* d_in, const int* in_sizes, int n_in,
                              void* d_out, int out_size, void* d_ws, size_t ws_size,
                              hipStream_t stream)
{
  (void)in_sizes; (void)n_in; (void)out_size; (void)ws_size;
  const float* x   = (const float*)d_in[0];
  const float* h   = (const float*)d_in[1];
  const float* qw1 = (const float*)d_in[2];
  const float* qb1 = (const float*)d_in[3];
  const float* qw2 = (const float*)d_in[4];
  const float* qb2 = (const float*)d_in[5];
  const float* kw1 = (const float*)d_in[6];
  const float* kb1 = (const float*)d_in[7];
  const float* kw2 = (const float*)d_in[8];
  const float* kb2 = (const float*)d_in[9];
  const float* vw1 = (const float*)d_in[10];
  const float* vb1 = (const float*)d_in[11];
  const float* vw2 = (const float*)d_in[12];
  const float* vb2 = (const float*)d_in[13];
  const float* xw1 = (const float*)d_in[14];
  const float* xb1 = (const float*)d_in[15];
  const float* xw2 = (const float*)d_in[16];
  const float* xb2 = (const float*)d_in[17];
  const float* how = (const float*)d_in[18];
  const float* hob = (const float*)d_in[19];
  const float* Wh  = (const float*)d_in[20];
  const float* Wu  = (const float*)d_in[21];

  float* out  = (float*)d_out;
  float* xout = out;                               // [B,3,T]
  float* hout = out + (size_t)B_ * 3 * T_;         // [B,D,T]

  // Workspace: qfrag 2MB | kvfrag 6MB | hupd 4MB | xacc 1MB | packed W 388KB
  char*  ws     = (char*)d_ws;
  uint4* qfrag  = (uint4*)(ws);
  uint4* kvfrag = (uint4*)(ws + (size_t)(2  << 20));
  float* hupd   = (float*)(ws + (size_t)(8  << 20));
  float* xacc   = (float*)(ws + (size_t)(12 << 20));
  char*  pwbase = ws + (size_t)(13 << 20);
  uint4* pwq1 = (uint4*)(pwbase);                  // (256,128): 64KB each
  uint4* pwk1 = (uint4*)(pwbase + (64 << 10));
  uint4* pwv1 = (uint4*)(pwbase + (128 << 10));
  uint4* pwq2 = (uint4*)(pwbase + (192 << 10));    // (128,128): 32KB each
  uint4* pwk2 = (uint4*)(pwbase + (224 << 10));
  uint4* pwv2 = (uint4*)(pwbase + (256 << 10));
  uint4* pwx1 = (uint4*)(pwbase + (288 << 10));
  uint4* pwx2 = (uint4*)(pwbase + (320 << 10));    // (128,16): 4KB
  uint4* pwho = (uint4*)(pwbase + (324 << 10));    // (128,256): 64KB

  et_packw_kernel<<<8, 256, 0, stream>>>(qw1, 256, 128, 128, pwq1);
  et_packw_kernel<<<8, 256, 0, stream>>>(kw1, 256, 128, 128, pwk1);
  et_packw_kernel<<<8, 256, 0, stream>>>(vw1, 256, 128, 128, pwv1);
  et_packw_kernel<<<4, 256, 0, stream>>>(qw2, 128, 128, 128, pwq2);
  et_packw_kernel<<<4, 256, 0, stream>>>(kw2, 128, 128, 128, pwk2);
  et_packw_kernel<<<4, 256, 0, stream>>>(vw2, 128, 128, 128, pwv2);
  et_packw_kernel<<<4, 256, 0, stream>>>(xw1, 128, 128, 128, pwx1);
  et_packw_kernel<<<1, 256, 0, stream>>>(xw2, 128,   8,  16, pwx2);
  et_packw_kernel<<<8, 256, 0, stream>>>(how, 128, 256, 256, pwho);

  et_qkv_kernel<<<512, 256, 0, stream>>>(
      x, h, qb1, qb2, kb1, kb2, vb1, vb2, xb1, xb2,
      pwq1, pwq2, pwk1, pwk2, pwv1, pwv2, pwx1, pwx2, qfrag, kvfrag);

  et_attn_kernel<<<512, 256, 0, stream>>>(qfrag, kvfrag, hupd, xacc);

  et_hout_kernel<<<512, 256, 0, stream>>>(h, hupd, pwho, hob, hout);

  et_xout_kernel<<<BT_ / 256, 256, 0, stream>>>(x, xacc, Wh, Wu, xout);
}
